// Policy_RNN_continous_60069412601899
// MI455X (gfx1250) — matrix operand order repaired and checked by reading, not compiled, not run on hardware
//
#include <hip/hip_runtime.h>
#include <hip/hip_bf16.h>
#include <math.h>

typedef __attribute__((ext_vector_type(16))) _Float16 v16h;
typedef __attribute__((ext_vector_type(8)))  float    v8f;

#define EPS_BN 1e-5f
#define PI_F   3.14159265358979323846f

// =====================================================================
// Kernel A: block0 fused  (depthwise 3x3 s2 + BN + ReLU  ->  pointwise
// 64->10 via WMMA f16 + BN + ReLU).  Grid: (13 tiles of 64 positions,
// 256 batches), 256 threads (8 waves).  This kernel streams the 205 MB
// input exactly once -> bandwidth-bound at ~9 us.
// =====================================================================
__global__ __launch_bounds__(256) void k_block0(
    const float* __restrict__ x,
    const float* __restrict__ dww, const float* __restrict__ pww,
    const float* __restrict__ g1, const float* __restrict__ b1p,
    const float* __restrict__ m1, const float* __restrict__ v1,
    const float* __restrict__ g2, const float* __restrict__ b2p,
    const float* __restrict__ m2, const float* __restrict__ v2,
    float* __restrict__ y1)
{
  __shared__ _Float16 dwA[64][72];   // [pos_local][channel], padded stride
  __shared__ _Float16 wB[64][16];    // [k][n]  (pointwise weight^T, N padded)
  __shared__ float wdw[576];         // 64ch x 9 depthwise weights
  __shared__ float s1[64], o1[64], s2[16], o2[16];

  const int tid  = threadIdx.x;
  const int b    = blockIdx.y;
  const int tile = blockIdx.x;       // 0..12, 64 positions each (784 total)

  for (int i = tid; i < 576; i += 256) wdw[i] = dww[i];
  if (tid < 64) {
    float sc = g1[tid] / sqrtf(v1[tid] + EPS_BN);
    s1[tid] = sc; o1[tid] = b1p[tid] - m1[tid] * sc;
  }
  if (tid < 16) {
    if (tid < 10) {
      float sc = g2[tid] / sqrtf(v2[tid] + EPS_BN);
      s2[tid] = sc; o2[tid] = b2p[tid] - m2[tid] * sc;
    } else { s2[tid] = 0.f; o2[tid] = 0.f; }
  }
  for (int i = tid; i < 64 * 16; i += 256) {
    int k = i >> 4, n = i & 15;
    wB[k][n] = (n < 10) ? (_Float16)pww[n * 64 + k] : (_Float16)0.f;
  }
  __syncthreads();

  // Phase 1: depthwise 3x3 stride2 pad1 + BN1 + ReLU -> LDS (f16)
  // idx = pos_local + 64*c; consecutive threads walk consecutive output
  // columns -> stride-2 coalesced reads of x.
  for (int it = 0; it < 16; ++it) {
    int idx = tid + it * 256;
    int c = idx >> 6, pl = idx & 63;
    int pos = tile * 64 + pl;
    _Float16 r = (_Float16)0.f;
    if (pos < 784) {
      int oy = pos / 28, ox = pos % 28;
      float acc = 0.f;
      #pragma unroll
      for (int ky = 0; ky < 3; ++ky) {
        int iy = 2 * oy - 1 + ky;
        if (iy < 0 || iy >= 56) continue;
        #pragma unroll
        for (int kx = 0; kx < 3; ++kx) {
          int ix = 2 * ox - 1 + kx;
          if (ix < 0 || ix >= 56) continue;
          acc += x[((b * 64 + c) * 56 + iy) * 56 + ix] * wdw[c * 9 + ky * 3 + kx];
        }
      }
      float v = acc * s1[c] + o1[c];
      r = (_Float16)(v > 0.f ? v : 0.f);
    }
    dwA[pl][c] = r;
  }
  __syncthreads();

  // Phase 2: pointwise GEMM [16 pos x 64 ch] x [64 ch x 16 out] via two
  // v_wmma_f32_16x16x32_f16 (K=64 = 2x32).  Waves 0..3 take one 16-row
  // block each (branch is wave-uniform -> EXEC all ones inside).
  const int wave = tid >> 5;
  if (wave < 4) {
    const int lane = tid & 31;
    const int m   = lane & 15;
    const int sel = lane >> 4;            // K-half selector per ISA layout
    const int rb  = wave;
    const int row = rb * 16 + m;
    v16h a0, a1, bb0, bb1;
    #pragma unroll
    for (int j = 0; j < 8; ++j) {
      // 16-bit A 16x32 layout: lanes 0-15: K 0..7 then 16..23; lanes 16-31:
      // K 8..15 then 24..31  (CDNA5 ISA §7.12.2)
      a0[j]     = dwA[row][ 0 + sel * 8 + j];
      a0[8 + j] = dwA[row][16 + sel * 8 + j];
      a1[j]     = dwA[row][32 + sel * 8 + j];
      a1[8 + j] = dwA[row][48 + sel * 8 + j];
    }
    const int n = lane & 15;
    #pragma unroll
    for (int j = 0; j < 16; ++j) {
      // B 32x16 layout: lane = column N; lanes 0-15 hold K 0..7 then 16..23,
      // lanes 16-31 hold K 8..15 then 24..31 (matches A).
      const int kk = (j < 8) ? (sel * 8 + j) : (16 + sel * 8 + (j - 8));
      bb0[j] = wB[ 0 + kk][n];
      bb1[j] = wB[32 + kk][n];
    }
    v8f c8 = {};
    c8 = __builtin_amdgcn_wmma_f32_16x16x32_f16(false, a0, false, bb0,
                                                (short)0, c8, false, false);
    c8 = __builtin_amdgcn_wmma_f32_16x16x32_f16(false, a1, false, bb1,
                                                (short)0, c8, false, false);
    // Epilogue: C/D layout -> lane n = N, VGPR r -> M = r + sel*8
    if (n < 10) {
      float sc = s2[n], of = o2[n];
      #pragma unroll
      for (int r = 0; r < 8; ++r) {
        int M = r + sel * 8;
        int pos = tile * 64 + rb * 16 + M;
        if (pos < 784) {
          float v = c8[r] * sc + of;
          y1[(b * 10 + n) * 784 + pos] = v > 0.f ? v : 0.f;
        }
      }
    }
  }
}

// =====================================================================
// Kernel B: blocks 1..3 + global avg pool, one workgroup per batch,
// everything staged in LDS (total live data per batch < 48 KB).
// =====================================================================
struct BParams { const float* p[30]; };

__device__ void ds_block(const float* gin, float* tmp, float* outb, int H,
                         const float* dww, const float* pww,
                         const float* g1, const float* b1, const float* m1, const float* v1,
                         const float* g2, const float* b2, const float* m2, const float* v2,
                         int tid)
{
  const int HO = (H + 1) / 2;
  const int NP = HO * HO;
  for (int i = tid; i < 10 * NP; i += 256) {
    int c = i / NP, p = i % NP;
    int oy = p / HO, ox = p % HO;
    float acc = 0.f;
    for (int ky = 0; ky < 3; ++ky) {
      int iy = 2 * oy - 1 + ky;
      if (iy < 0 || iy >= H) continue;
      for (int kx = 0; kx < 3; ++kx) {
        int ix = 2 * ox - 1 + kx;
        if (ix < 0 || ix >= H) continue;
        acc += gin[(c * H + iy) * H + ix] * dww[c * 9 + ky * 3 + kx];
      }
    }
    float sc = g1[c] / sqrtf(v1[c] + EPS_BN);
    float v = acc * sc + (b1[c] - m1[c] * sc);
    tmp[i] = v > 0.f ? v : 0.f;
  }
  __syncthreads();
  for (int i = tid; i < 10 * NP; i += 256) {
    int co = i / NP, p = i % NP;
    float acc = 0.f;
    for (int ci = 0; ci < 10; ++ci) acc += tmp[ci * NP + p] * pww[co * 10 + ci];
    float sc = g2[co] / sqrtf(v2[co] + EPS_BN);
    float v = acc * sc + (b2[co] - m2[co] * sc);
    outb[i] = v > 0.f ? v : 0.f;
  }
  __syncthreads();
}

__global__ __launch_bounds__(256) void k_tail(const float* __restrict__ y1,
                                              BParams bp,
                                              float* __restrict__ feat)
{
  __shared__ float bufA[7840];   // [10][28][28] then reused
  __shared__ float bufT[1960];
  __shared__ float bufB[1960];
  const int tid = threadIdx.x, b = blockIdx.x;
  for (int i = tid; i < 7840; i += 256) bufA[i] = y1[b * 7840 + i];
  __syncthreads();
  ds_block(bufA, bufT, bufB, 28,
           bp.p[0], bp.p[1], bp.p[2], bp.p[3], bp.p[4], bp.p[5],
           bp.p[6], bp.p[7], bp.p[8], bp.p[9], tid);            // b1: 28->14
  ds_block(bufB, bufT, bufA, 14,
           bp.p[10], bp.p[11], bp.p[12], bp.p[13], bp.p[14], bp.p[15],
           bp.p[16], bp.p[17], bp.p[18], bp.p[19], tid);        // b2: 14->7
  ds_block(bufA, bufT, bufB, 7,
           bp.p[20], bp.p[21], bp.p[22], bp.p[23], bp.p[24], bp.p[25],
           bp.p[26], bp.p[27], bp.p[28], bp.p[29], tid);        // b3: 7->4
  if (tid < 10) {
    float s = 0.f;
    for (int i = 0; i < 16; ++i) s += bufB[tid * 16 + i];
    feat[b * 10 + tid] = s * (1.f / 16.f);
  }
}

// =====================================================================
// Kernel C: LSTM policy scan.  One wave32 per batch element (8 per
// block), lanes parallelize the 40 gate dot-products; lanes 0/1 run the
// mu/var heads, sampling, log-prob/entropy and the bits recurrence.
// =====================================================================
struct LParams { const float* p[12]; };

__device__ inline float sigmoidf_(float x) { return 1.f / (1.f + expf(-x)); }
__device__ inline float softplusf_(float x) { return (x > 20.f) ? x : log1pf(expf(x)); }

__global__ __launch_bounds__(256) void k_lstm(const float* __restrict__ feat,
                                              const float* __restrict__ eps_a,
                                              const float* __restrict__ eps_w,
                                              LParams lp,
                                              float* __restrict__ out)
{
  __shared__ float fs[8][10], h1[8][10], c1[8][10], h2[8][10], c2[8][10];
  __shared__ float sub[8][2], gb[8][40];
  const int tid = threadIdx.x, w = tid >> 5, lane = tid & 31;
  const int b = blockIdx.x * 8 + w;
  const float* wih0 = lp.p[0]; const float* whh0 = lp.p[1];
  const float* bih0 = lp.p[2]; const float* bhh0 = lp.p[3];
  const float* wih1 = lp.p[4]; const float* whh1 = lp.p[5];
  const float* bih1 = lp.p[6]; const float* bhh1 = lp.p[7];
  const float* wmu  = lp.p[8]; const float* bmu  = lp.p[9];
  const float* wvar = lp.p[10]; const float* bvar = lp.p[11];

  if (lane < 10) {
    fs[w][lane] = feat[b * 10 + lane];
    h1[w][lane] = 0.f; c1[w][lane] = 0.f; h2[w][lane] = 0.f; c2[w][lane] = 0.f;
  }
  if (lane < 2) sub[w][lane] = 0.f;
  __syncthreads();

  for (int t = 0; t < 19; ++t) {
    // ---- cell 1 gates (input = [feat(10), subseq(2)]) ----
    for (int pass = 0; pass < 2; ++pass) {
      int g = lane + pass * 32;
      if (g < 40) {
        float acc = bih0[g] + bhh0[g];
        for (int j = 0; j < 10; ++j) acc += wih0[g * 12 + j] * fs[w][j];
        acc += wih0[g * 12 + 10] * sub[w][0];
        acc += wih0[g * 12 + 11] * sub[w][1];
        for (int j = 0; j < 10; ++j) acc += whh0[g * 10 + j] * h1[w][j];
        gb[w][g] = acc;
      }
    }
    __syncthreads();
    if (lane < 10) {
      float ig = sigmoidf_(gb[w][lane]);
      float fg = sigmoidf_(gb[w][10 + lane]);
      float gg = tanhf(gb[w][20 + lane]);
      float og = sigmoidf_(gb[w][30 + lane]);
      float c = fg * c1[w][lane] + ig * gg;
      c1[w][lane] = c;
      h1[w][lane] = og * tanhf(c);
    }
    __syncthreads();
    // ---- cell 2 gates (input = h1) ----
    for (int pass = 0; pass < 2; ++pass) {
      int g = lane + pass * 32;
      if (g < 40) {
        float acc = bih1[g] + bhh1[g];
        for (int j = 0; j < 10; ++j) acc += wih1[g * 10 + j] * h1[w][j];
        for (int j = 0; j < 10; ++j) acc += whh1[g * 10 + j] * h2[w][j];
        gb[w][g] = acc;
      }
    }
    __syncthreads();
    if (lane < 10) {
      float ig = sigmoidf_(gb[w][lane]);
      float fg = sigmoidf_(gb[w][10 + lane]);
      float gg = tanhf(gb[w][20 + lane]);
      float og = sigmoidf_(gb[w][30 + lane]);
      float c = fg * c2[w][lane] + ig * gg;
      c2[w][lane] = c;
      h2[w][lane] = og * tanhf(c);
    }
    __syncthreads();
    // ---- heads + sampling (lane 0 -> a-channel, lane 1 -> w-channel) ----
    if (lane < 2) {
      int k = lane;
      float am = bmu[k], av = bvar[k];
      for (int j = 0; j < 10; ++j) {
        am += wmu[k * 10 + j] * h2[w][j];
        av += wvar[k * 10 + j] * h2[w][j];
      }
      float mu  = softplusf_(am);
      float var = softplusf_(av);
      float ep  = (k == 0 ? eps_a : eps_w)[t * 256 + b];
      float act = mu + sqrtf(var) * ep;
      float d   = act - mu;
      float lpv = -d * d / (2.f * var) - 0.5f * logf(2.f * var * PI_F);
      float ent = -0.5f * (logf(var + 2.f * PI_F) + 1.f);
      float ac  = fminf(fmaxf(act, 0.f), 6.f);
      float bits = rintf(ac);  // round-half-even like jnp.round
      out[(0 + k) * 4864 + b * 19 + t] = bits + 2.f;  // bit_a / bit_w
      out[(2 + k) * 4864 + b * 19 + t] = lpv;         // lpa / lpw
      out[(4 + k) * 4864 + b * 19 + t] = ent;         // ea / ew
      sub[w][k] = bits;
    }
    __syncthreads();
  }
}

// =====================================================================
extern "C" void kernel_launch(void* const* d_in, const int* in_sizes, int n_in,
                              void* d_out, int out_size, void* d_ws, size_t ws_size,
                              hipStream_t stream) {
  (void)in_sizes; (void)n_in; (void)out_size; (void)ws_size;
  const float* x     = (const float*)d_in[0];
  const float* eps_a = (const float*)d_in[1];
  const float* eps_w = (const float*)d_in[2];
  auto P = [&](int i) { return (const float*)d_in[3 + i]; };

  float* y1   = (float*)d_ws;            // [256,10,784] = 8.03 MB
  float* feat = y1 + 256 * 10 * 784;     // [256,10]

  // block0: dw+pw fused, WMMA pointwise.  13 tiles x 256 batches.
  dim3 gA(13, 256);
  k_block0<<<gA, 256, 0, stream>>>(x, P(0), P(1), P(2), P(3), P(4), P(5),
                                   P(6), P(7), P(8), P(9), y1);

  // blocks 1..3 + avgpool
  BParams bp;
  for (int blk = 0; blk < 3; ++blk)
    for (int j = 0; j < 10; ++j)
      bp.p[blk * 10 + j] = P((blk + 1) * 10 + j);
  k_tail<<<256, 256, 0, stream>>>(y1, bp, feat);

  // LSTM policy scan
  LParams lpar;
  for (int j = 0; j < 12; ++j) lpar.p[j] = P(40 + j);
  k_lstm<<<32, 256, 0, stream>>>(feat, eps_a, eps_w, lpar, (float*)d_out);
}
